// SwinCrossViewFusion_2662879723765
// MI455X (gfx1250) — compile-verified, matrix operand order checked by reading, not run on hardware
//
// MI455X (gfx1250) Swin-v2 cross-view fusion.
// - All large matmuls: V_WMMA_F32_16X16X32_BF16, fp32 accumulation.
// - Weights pre-converted to bf16 once (halves L2 re-read traffic; fc1/fc2
//   become L2-resident at 67MB each vs 134MB fp32).
// - GEMM B-tiles staged by the Tensor Data Mover (tensor_load_to_lds, 2-D
//   descriptor, pad_enable for a 34-half LDS row stride), double-buffered and
//   overlapped with WMMA; A-tiles staged as float4 loads + bf16 convert.
// - Fused window attention per (window, head) workgroup, also WMMA.
#include <hip/hip_runtime.h>
#include <hip/hip_bf16.h>

typedef __bf16 bf16_t;
typedef bf16_t v16bf __attribute__((ext_vector_type(16)));
typedef float  v8f   __attribute__((ext_vector_type(8)));
typedef unsigned int u32x4 __attribute__((ext_vector_type(4)));
typedef int i32x8 __attribute__((ext_vector_type(8)));
typedef int i32x4 __attribute__((ext_vector_type(4)));

#define C_DIM   2048
#define NHEAD   16
#define HDIM    128
#define L_TOK   588
#define BATCH   8
#define M_TOK   (BATCH * L_TOK)      // 4704
#define DEPTH   4
#define WINSZ   7
#define NTOK    49
#define SHIFT_A 3
#define HGRID   42
#define WGRID   14
#define NWIN    12
#define M_FUSE  (BATCH * 196)        // 1568

#define BK 32

__device__ __forceinline__ float gelu_exact(float v) {
  return 0.5f * v * (1.0f + erff(v * 0.70710678118654752f));
}

// ---------------------------------------------------------------------------
// bf16 WMMA GEMM:  Y[M,N] = X[M,K](f32) @ Wb[N,K](bf16)^T + bias[N]
// 128x128 block tile, BK=32, double-buffered LDS.
// B tiles: TDM async DMA (wave 0 issues, TENSORcnt-waited).
// A tiles: float4 loads + convert, all 8 waves.
// ---------------------------------------------------------------------------
template <int EPI>
__global__ __launch_bounds__(256) void gemm_bf16_kernel(
    const float* __restrict__ X, const bf16_t* __restrict__ Wb,
    const float* __restrict__ bias, float* __restrict__ Y,
    int M, int N, int K) {
  __shared__ bf16_t sA[2][128][36];
  __shared__ bf16_t sB[2][128][34];  // TDM writes 32 bf16 + 1 pad dword / row

  const int tid  = threadIdx.x;
  const int lane = tid & 31;
  const int wave = tid >> 5;
  const int wm   = wave >> 1;     // 0..3 : 32 rows
  const int wn   = wave & 1;      // 0..1 : 64 cols
  const int bm0  = blockIdx.y * 128;
  const int bn0  = blockIdx.x * 128;

  const int mlane = lane & 15;
  const int kbase = (lane >> 4) * 8;

  // TDM descriptor issue: load bf16 tile [128 rows x 32 cols] of Wb starting
  // at row bn0, col k0 into sB[buf]. data_size=2B (code 1), pad 1 dword per
  // 16 stored dwords (=64B row) -> LDS row stride 68B = 34 halves.
  auto issue_tdm = [&](int buf, int k0) {
    unsigned int lds_off = (unsigned int)(size_t)(&sB[buf][0][0]);
    unsigned long long ga =
        (unsigned long long)(size_t)(Wb + (size_t)bn0 * K + k0);
    u32x4 g0;
    g0[0] = 1u;                                  // count=1, user descriptor
    g0[1] = lds_off;                             // lds_addr
    g0[2] = (unsigned int)(ga & 0xffffffffull);  // global_addr[31:0]
    g0[3] = (unsigned int)((ga >> 32) & 0x1ffffffull) | (2u << 30);  // type=2
    i32x8 g1;
    g1[0] = (int)((1u << 16)      // data_size = 2 bytes
                  | (1u << 20)    // pad_enable
                  | (3u << 22));  // pad_interval = 16 dwords (64B)
                                  // pad_amount = 0 -> 1 dword
    g1[1] = (int)(((unsigned int)K & 0xffffu) << 16);           // tdim0 lo
    g1[2] = (int)(((unsigned int)K >> 16) |
                  (((unsigned int)N & 0xffffu) << 16));         // tdim0 hi|td1
    g1[3] = (int)(((unsigned int)N >> 16) | (32u << 16));       // tile_dim0=32
    g1[4] = 128;                                                // tile_dim1
    g1[5] = K;                                                  // dim0 stride
    g1[6] = 0;
    g1[7] = 0;
    i32x4 z4 = (i32x4)0;
    i32x8 z8 = (i32x8)0;
    __builtin_amdgcn_tensor_load_to_lds(g0, g1, z4, z4, z8, 0);
  };

  // A tile stage: 128x32 fp32 -> bf16, float4-vectorized, row-guarded.
  auto stage_a = [&](int buf, int k0) {
#pragma unroll
    for (int i = 0; i < 4; ++i) {
      int e = (i * 256 + tid) * 4;
      int r = e >> 5, c = e & 31;
      int gr = bm0 + r;
      float4 v = make_float4(0.f, 0.f, 0.f, 0.f);
      if (gr < M)
        v = *reinterpret_cast<const float4*>(X + (size_t)gr * K + k0 + c);
      sA[buf][r][c + 0] = (bf16_t)v.x;
      sA[buf][r][c + 1] = (bf16_t)v.y;
      sA[buf][r][c + 2] = (bf16_t)v.z;
      sA[buf][r][c + 3] = (bf16_t)v.w;
    }
  };

  v8f acc[2][4];
#pragma unroll
  for (int i = 0; i < 2; ++i)
#pragma unroll
    for (int j = 0; j < 4; ++j) acc[i][j] = (v8f)(0.0f);

  // Prologue: fill buffer 0.
  if (tid < 32) issue_tdm(0, 0);
  stage_a(0, 0);
  if (tid < 32) __builtin_amdgcn_s_wait_tensorcnt(0);
  __syncthreads();

  int cur = 0;
  for (int k0 = 0; k0 < K; k0 += BK) {
    const int nxt = cur ^ 1;
    const bool more = (k0 + BK) < K;
    if (more) {
      if (tid < 32) issue_tdm(nxt, k0 + BK);
      stage_a(nxt, k0 + BK);
      __builtin_prefetch(&X[(size_t)(bm0 + (tid >> 1)) * K + (k0 + 2 * BK)],
                         0, 1);
    }

    v16bf afrag[2], bfrag[4];
#pragma unroll
    for (int mt = 0; mt < 2; ++mt) {
      int row = wm * 32 + mt * 16 + mlane;
#pragma unroll
      for (int j = 0; j < 16; ++j) {
        int kk = kbase + j + ((j >= 8) ? 8 : 0);
        afrag[mt][j] = sA[cur][row][kk];
      }
    }
#pragma unroll
    for (int nt = 0; nt < 4; ++nt) {
      int col = wn * 64 + nt * 16 + mlane;
#pragma unroll
      for (int j = 0; j < 16; ++j) {
        int kk = kbase + j + ((j >= 8) ? 8 : 0);
        bfrag[nt][j] = sB[cur][col][kk];
      }
    }
#pragma unroll
    for (int mt = 0; mt < 2; ++mt)
#pragma unroll
      for (int nt = 0; nt < 4; ++nt)
        acc[mt][nt] = __builtin_amdgcn_wmma_f32_16x16x32_bf16(
            false, afrag[mt], false, bfrag[nt], (short)0, acc[mt][nt], false,
            false);

    if (more && tid < 32) __builtin_amdgcn_s_wait_tensorcnt(0);
    __syncthreads();
    cur = nxt;
  }

  const int rbase = (lane >> 4) * 8;
  const int cl    = lane & 15;
#pragma unroll
  for (int mt = 0; mt < 2; ++mt)
#pragma unroll
    for (int nt = 0; nt < 4; ++nt)
#pragma unroll
      for (int r = 0; r < 8; ++r) {
        int grow = bm0 + wm * 32 + mt * 16 + rbase + r;
        int gcol = bn0 + wn * 64 + nt * 16 + cl;
        if (grow < M) {
          float v = acc[mt][nt][r] + bias[gcol];
          if (EPI == 1) v = gelu_exact(v);
          Y[(size_t)grow * N + gcol] = v;
        }
      }
}

// ---------------------------------------------------------------------------
// Fused window attention (one workgroup per (window, head)).
// ---------------------------------------------------------------------------
__device__ __forceinline__ int region_id(int h, int w) {
  int rh = (h < HGRID - WINSZ) ? 0 : ((h < HGRID - SHIFT_A) ? 1 : 2);
  int rw = (w < WGRID - WINSZ) ? 0 : ((w < WGRID - SHIFT_A) ? 1 : 2);
  return rh * 3 + rw;
}

__global__ __launch_bounds__(256) void attn_kernel(
    const float* __restrict__ qkv, const float* __restrict__ biasTab,
    const float* __restrict__ lscale, float* __restrict__ out, int shift) {
  __shared__ bf16_t sQb[64][132];
  __shared__ bf16_t sKb[64][132];
  __shared__ float  sS[64][66];
  __shared__ int    sSrc[NTOK];

  const int blk  = blockIdx.x;
  const int head = blk & (NHEAD - 1);
  const int win  = blk >> 4;
  const int b    = win / NWIN;
  const int w12  = win % NWIN;
  const int wh   = w12 >> 1;
  const int ww   = w12 & 1;

  const int tid  = threadIdx.x;
  const int lane = tid & 31;
  const int wave = tid >> 5;

  for (int idx = tid; idx < 64 * 132; idx += 256) {
    sQb[idx / 132][idx % 132] = (bf16_t)0.0f;
    sKb[idx / 132][idx % 132] = (bf16_t)0.0f;
  }
  for (int idx = tid; idx < 64 * 66; idx += 256) sS[idx / 66][idx % 66] = 0.0f;
  if (tid < NTOK) {
    int th = tid / WINSZ, tw = tid % WINSZ;
    int h = wh * WINSZ + th, w = ww * WINSZ + tw;
    int hs = h + shift; if (hs >= HGRID) hs -= HGRID;
    int ws = w + shift; if (ws >= WGRID) ws -= WGRID;
    sSrc[tid] = hs * WGRID + ws;
  }
  __syncthreads();

  const float lsc   = lscale[head];
  const float scale = __expf(fminf(lsc, 4.605170186f));  // ln(100)
  const int qoff = head * HDIM;
  const int koff = C_DIM + head * HDIM;
  for (int r = wave; r < NTOK; r += 8) {
    size_t row = (size_t)(b * L_TOK + sSrc[r]) * (3 * C_DIM);
    float q0 = qkv[row + qoff + lane],      q1 = qkv[row + qoff + lane + 32];
    float q2 = qkv[row + qoff + lane + 64], q3 = qkv[row + qoff + lane + 96];
    float k0 = qkv[row + koff + lane],      k1 = qkv[row + koff + lane + 32];
    float k2 = qkv[row + koff + lane + 64], k3 = qkv[row + koff + lane + 96];
    float sq = q0 * q0 + q1 * q1 + q2 * q2 + q3 * q3;
    float sk = k0 * k0 + k1 * k1 + k2 * k2 + k3 * k3;
#pragma unroll
    for (int o = 16; o > 0; o >>= 1) {
      sq += __shfl_xor(sq, o, 32);
      sk += __shfl_xor(sk, o, 32);
    }
    float qs = rsqrtf(sq + 1e-12f) * scale;
    float ks = rsqrtf(sk + 1e-12f);
    sQb[r][lane]      = (bf16_t)(q0 * qs);
    sQb[r][lane + 32] = (bf16_t)(q1 * qs);
    sQb[r][lane + 64] = (bf16_t)(q2 * qs);
    sQb[r][lane + 96] = (bf16_t)(q3 * qs);
    sKb[r][lane]      = (bf16_t)(k0 * ks);
    sKb[r][lane + 32] = (bf16_t)(k1 * ks);
    sKb[r][lane + 64] = (bf16_t)(k2 * ks);
    sKb[r][lane + 96] = (bf16_t)(k3 * ks);
  }
  __syncthreads();

  const int mlane = lane & 15;
  const int kbase = (lane >> 4) * 8;
  const int rbase = (lane >> 4) * 8;
  {  // S = q @ k^T
    int mt = wave >> 1;
    int nt0 = (wave & 1) * 2;
    v8f acc[2];
    acc[0] = (v8f)(0.0f);
    acc[1] = (v8f)(0.0f);
    for (int ks = 0; ks < HDIM; ks += 32) {
      v16bf afrag, bfrag0, bfrag1;
#pragma unroll
      for (int j = 0; j < 16; ++j) {
        int kk = ks + kbase + j + ((j >= 8) ? 8 : 0);
        afrag[j]  = sQb[mt * 16 + mlane][kk];
        bfrag0[j] = sKb[nt0 * 16 + mlane][kk];
        bfrag1[j] = sKb[(nt0 + 1) * 16 + mlane][kk];
      }
      acc[0] = __builtin_amdgcn_wmma_f32_16x16x32_bf16(false, afrag, false,
                                                       bfrag0, (short)0,
                                                       acc[0], false, false);
      acc[1] = __builtin_amdgcn_wmma_f32_16x16x32_bf16(false, afrag, false,
                                                       bfrag1, (short)0,
                                                       acc[1], false, false);
    }
#pragma unroll
    for (int t = 0; t < 2; ++t)
#pragma unroll
      for (int r = 0; r < 8; ++r)
        sS[mt * 16 + rbase + r][(nt0 + t) * 16 + mlane] = acc[t][r];
  }
  __syncthreads();

  if (tid < 64) {  // softmax (+CPB bias, +shift mask)
    int r = tid;
    if (r < NTOK) {
      int ih = wh * WINSZ + r / WINSZ, iw = ww * WINSZ + r % WINSZ;
      int regI = region_id(ih, iw);
      const float* brow = biasTab + ((size_t)head * NTOK + r) * NTOK;
      float mx = -1e30f;
      for (int j = 0; j < NTOK; ++j) {
        float v = sS[r][j] + brow[j];
        if (shift) {
          int jh = wh * WINSZ + j / WINSZ, jw = ww * WINSZ + j % WINSZ;
          if (region_id(jh, jw) != regI) v -= 100.0f;
        }
        mx = fmaxf(mx, v);
      }
      float sum = 0.0f;
      for (int j = 0; j < NTOK; ++j) {
        float v = sS[r][j] + brow[j];
        if (shift) {
          int jh = wh * WINSZ + j / WINSZ, jw = ww * WINSZ + j % WINSZ;
          if (region_id(jh, jw) != regI) v -= 100.0f;
        }
        float e = __expf(v - mx);
        sS[r][j] = e;
        sum += e;
      }
      float inv = 1.0f / sum;
      for (int j = 0; j < NTOK; ++j) sS[r][j] *= inv;
      for (int j = NTOK; j < 64; ++j) sS[r][j] = 0.0f;
    } else {
      for (int j = 0; j < 64; ++j) sS[r][j] = 0.0f;
    }
  }
  __syncthreads();

  {  // O = P @ V
    int mt  = wave >> 1;
    int nt0 = (wave & 1) * 4;
    const int voff = 2 * C_DIM + head * HDIM;
    v8f acc[4];
#pragma unroll
    for (int t = 0; t < 4; ++t) acc[t] = (v8f)(0.0f);
    for (int ks = 0; ks < 64; ks += 32) {
      v16bf afrag;
#pragma unroll
      for (int j = 0; j < 16; ++j) {
        int kk = ks + kbase + j + ((j >= 8) ? 8 : 0);
        afrag[j] = (bf16_t)sS[mt * 16 + mlane][kk];
      }
#pragma unroll
      for (int t = 0; t < 4; ++t) {
        int c = (nt0 + t) * 16 + mlane;
        v16bf bfrag;
#pragma unroll
        for (int j = 0; j < 16; ++j) {
          int kk = ks + kbase + j + ((j >= 8) ? 8 : 0);
          float v = 0.0f;
          if (kk < NTOK)
            v = qkv[(size_t)(b * L_TOK + sSrc[kk]) * (3 * C_DIM) + voff + c];
          bfrag[j] = (bf16_t)v;
        }
        acc[t] = __builtin_amdgcn_wmma_f32_16x16x32_bf16(false, afrag, false,
                                                         bfrag, (short)0,
                                                         acc[t], false, false);
      }
    }
#pragma unroll
    for (int t = 0; t < 4; ++t)
#pragma unroll
      for (int r = 0; r < 8; ++r) {
        int tok = mt * 16 + rbase + r;
        if (tok < NTOK) {
          int c = (nt0 + t) * 16 + mlane;
          out[(size_t)(b * L_TOK + sSrc[tok]) * C_DIM + head * HDIM + c] =
              acc[t][r];
        }
      }
  }
}

// ---------------------------------------------------------------------------
// Support kernels
// ---------------------------------------------------------------------------
__global__ void to_bf16_kernel(const float* __restrict__ src,
                               bf16_t* __restrict__ dst, int n) {
  int i = blockIdx.x * blockDim.x + threadIdx.x;
  if (i < n) dst[i] = (bf16_t)src[i];
}

__global__ void qkv_bias_kernel(const float* __restrict__ qb,
                                const float* __restrict__ vb,
                                float* __restrict__ dst) {
  int idx = blockIdx.x * blockDim.x + threadIdx.x;
  if (idx >= DEPTH * 3 * C_DIM) return;
  int d = idx / (3 * C_DIM);
  int j = idx % (3 * C_DIM);
  float v = 0.0f;
  if (j < C_DIM) v = qb[d * C_DIM + j];
  else if (j >= 2 * C_DIM) v = vb[d * C_DIM + (j - 2 * C_DIM)];
  dst[idx] = v;
}

__device__ __forceinline__ float cpb_coord(int r) {
  float a = fabsf((float)r) * (8.0f / 6.0f);
  float v = log2f(a + 1.0f) * (1.0f / 3.0f);
  return (r < 0) ? -v : v;
}

__global__ void cpb_kernel(const float* __restrict__ w1,
                           const float* __restrict__ b1,
                           const float* __restrict__ w2,
                           float* __restrict__ biasTab) {
  int idx = blockIdx.x * blockDim.x + threadIdx.x;
  if (idx >= DEPTH * NHEAD * NTOK * NTOK) return;
  int j = idx % NTOK;
  int i = (idx / NTOK) % NTOK;
  int h = (idx / (NTOK * NTOK)) % NHEAD;
  int d = idx / (NTOK * NTOK * NHEAD);
  int dh = (i / WINSZ) - (j / WINSZ);
  int dw = (i % WINSZ) - (j % WINSZ);
  float t0 = cpb_coord(dh), t1 = cpb_coord(dw);
  const float* w1d = w1 + (size_t)d * 512 * 2;
  const float* b1d = b1 + (size_t)d * 512;
  const float* w2d = w2 + ((size_t)d * NHEAD + h) * 512;
  float acc = 0.0f;
  for (int k = 0; k < 512; ++k) {
    float hv = w1d[k * 2] * t0 + w1d[k * 2 + 1] * t1 + b1d[k];
    hv = fmaxf(hv, 0.0f);
    acc += hv * w2d[k];
  }
  biasTab[idx] = 16.0f / (1.0f + __expf(-acc));
}

__global__ void pack_kernel(const float* __restrict__ l0,
                            const float* __restrict__ l1,
                            const float* __restrict__ l2,
                            float* __restrict__ x) {
  int idx = blockIdx.x * blockDim.x + threadIdx.x;
  if (idx >= M_TOK * C_DIM) return;
  int c = idx % C_DIM;
  int l = (idx / C_DIM) % L_TOK;
  int b = idx / (C_DIM * L_TOK);
  int v = l / 196, p = l % 196;
  const float* src = (v == 0) ? l0 : ((v == 1) ? l1 : l2);
  x[idx] = src[((size_t)b * C_DIM + c) * 196 + p];
}

__global__ __launch_bounds__(256) void add_ln_kernel(
    float* __restrict__ x, const float* __restrict__ y,
    const float* __restrict__ g, const float* __restrict__ bt) {
  __shared__ float r1[256], r2[256];
  int row = blockIdx.x, tid = threadIdx.x;
  const float* yr = y + (size_t)row * C_DIM;
  float* xr = x + (size_t)row * C_DIM;
  float s = 0.0f, s2 = 0.0f;
  for (int c = tid; c < C_DIM; c += 256) {
    float v = yr[c];
    s += v;
    s2 += v * v;
  }
  r1[tid] = s; r2[tid] = s2;
  __syncthreads();
  for (int o = 128; o > 0; o >>= 1) {
    if (tid < o) { r1[tid] += r1[tid + o]; r2[tid] += r2[tid + o]; }
    __syncthreads();
  }
  float m = r1[0] * (1.0f / C_DIM);
  float var = r2[0] * (1.0f / C_DIM) - m * m;
  float inv = rsqrtf(var + 1e-5f);
  for (int c = tid; c < C_DIM; c += 256)
    xr[c] = xr[c] + (yr[c] - m) * inv * g[c] + bt[c];
}

__global__ void fuse_pack_kernel(const float* __restrict__ x,
                                 float* __restrict__ xf) {
  int idx = blockIdx.x * blockDim.x + threadIdx.x;
  if (idx >= M_FUSE * 3 * C_DIM) return;
  int col = idx % (3 * C_DIM);
  int row = idx / (3 * C_DIM);
  int b = row / 196, p = row % 196;
  int v = col / C_DIM, c = col % C_DIM;
  xf[idx] = x[((size_t)(b * L_TOK + v * 196 + p)) * C_DIM + c];
}

__global__ __launch_bounds__(256) void ln_out_kernel(
    const float* __restrict__ y, const float* __restrict__ g,
    const float* __restrict__ bt, float* __restrict__ out) {
  __shared__ float r1[256], r2[256];
  int row = blockIdx.x, tid = threadIdx.x;
  const float* yr = y + (size_t)row * C_DIM;
  int b = row / 196, p = row % 196;
  float s = 0.0f, s2 = 0.0f;
  for (int c = tid; c < C_DIM; c += 256) {
    float v = yr[c];
    s += v;
    s2 += v * v;
  }
  r1[tid] = s; r2[tid] = s2;
  __syncthreads();
  for (int o = 128; o > 0; o >>= 1) {
    if (tid < o) { r1[tid] += r1[tid + o]; r2[tid] += r2[tid + o]; }
    __syncthreads();
  }
  float m = r1[0] * (1.0f / C_DIM);
  float var = r2[0] * (1.0f / C_DIM) - m * m;
  float inv = rsqrtf(var + 1e-5f);
  for (int c = tid; c < C_DIM; c += 256)
    out[((size_t)b * C_DIM + c) * 196 + p] = (yr[c] - m) * inv * g[c] + bt[c];
}

// ---------------------------------------------------------------------------
extern "C" void kernel_launch(void* const* d_in, const int* in_sizes, int n_in,
                              void* d_out, int out_size, void* d_ws,
                              size_t ws_size, hipStream_t stream) {
  const float* lat0    = (const float*)d_in[0];
  const float* lat1    = (const float*)d_in[1];
  const float* lat2    = (const float*)d_in[2];
  const float* norm1_g = (const float*)d_in[3];
  const float* norm1_b = (const float*)d_in[4];
  const float* qkv_w   = (const float*)d_in[5];
  const float* q_bias  = (const float*)d_in[6];
  const float* v_bias  = (const float*)d_in[7];
  const float* lscale  = (const float*)d_in[8];
  const float* cpb_w1  = (const float*)d_in[9];
  const float* cpb_b1  = (const float*)d_in[10];
  const float* cpb_w2  = (const float*)d_in[11];
  const float* proj_w  = (const float*)d_in[12];
  const float* proj_b  = (const float*)d_in[13];
  const float* norm2_g = (const float*)d_in[14];
  const float* norm2_b = (const float*)d_in[15];
  const float* fc1_w   = (const float*)d_in[16];
  const float* fc1_b   = (const float*)d_in[17];
  const float* fc2_w   = (const float*)d_in[18];
  const float* fc2_b   = (const float*)d_in[19];
  const float* fus_w   = (const float*)d_in[20];
  const float* fus_b   = (const float*)d_in[21];
  const float* fn_g    = (const float*)d_in[22];
  const float* fn_b    = (const float*)d_in[23];

  float* xbuf    = (float*)d_ws;                          // [4704,2048]
  float* ybuf    = xbuf + (size_t)M_TOK * C_DIM;          // [4704,2048]
  float* tbuf    = ybuf + (size_t)M_TOK * C_DIM;          // [4704,6144]
  float* hbuf    = tbuf + (size_t)M_TOK * 3 * C_DIM;      // [4704,8192]
  float* biasTab = hbuf + (size_t)M_TOK * 4 * C_DIM;      // [4,16,49,49]
  float* qkvbias = biasTab + (size_t)DEPTH * NHEAD * NTOK * NTOK;  // [4,6144]
  // bf16 weight caches
  bf16_t* wqkv = (bf16_t*)(qkvbias + (size_t)DEPTH * 3 * C_DIM);
  bf16_t* wproj = wqkv + (size_t)DEPTH * 3 * C_DIM * C_DIM;
  bf16_t* wfc1 = wproj + (size_t)DEPTH * C_DIM * C_DIM;
  bf16_t* wfc2 = wfc1 + (size_t)DEPTH * 4 * C_DIM * C_DIM;
  bf16_t* wfus = wfc2 + (size_t)DEPTH * 4 * C_DIM * C_DIM;

  const int MB = (M_TOK + 127) / 128;  // 37

  // One-time bf16 weight conversion (graph-replayed; deterministic).
  {
    int n;
    n = DEPTH * 3 * C_DIM * C_DIM;
    to_bf16_kernel<<<(n + 255) / 256, 256, 0, stream>>>(qkv_w, wqkv, n);
    n = DEPTH * C_DIM * C_DIM;
    to_bf16_kernel<<<(n + 255) / 256, 256, 0, stream>>>(proj_w, wproj, n);
    n = DEPTH * 4 * C_DIM * C_DIM;
    to_bf16_kernel<<<(n + 255) / 256, 256, 0, stream>>>(fc1_w, wfc1, n);
    n = DEPTH * 4 * C_DIM * C_DIM;
    to_bf16_kernel<<<(n + 255) / 256, 256, 0, stream>>>(fc2_w, wfc2, n);
    n = 3 * C_DIM * C_DIM;
    to_bf16_kernel<<<(n + 255) / 256, 256, 0, stream>>>(fus_w, wfus, n);
  }

  qkv_bias_kernel<<<(DEPTH * 3 * C_DIM + 255) / 256, 256, 0, stream>>>(
      q_bias, v_bias, qkvbias);
  cpb_kernel<<<(DEPTH * NHEAD * NTOK * NTOK + 255) / 256, 256, 0, stream>>>(
      cpb_w1, cpb_b1, cpb_w2, biasTab);
  pack_kernel<<<(M_TOK * C_DIM + 255) / 256, 256, 0, stream>>>(lat0, lat1,
                                                               lat2, xbuf);

  for (int d = 0; d < DEPTH; ++d) {
    int shift = (d & 1) ? SHIFT_A : 0;
    gemm_bf16_kernel<0><<<dim3((3 * C_DIM) / 128, MB), 256, 0, stream>>>(
        xbuf, wqkv + (size_t)d * 3 * C_DIM * C_DIM, qkvbias + d * 3 * C_DIM,
        tbuf, M_TOK, 3 * C_DIM, C_DIM);
    attn_kernel<<<dim3(BATCH * NWIN * NHEAD), 256, 0, stream>>>(
        tbuf, biasTab + (size_t)d * NHEAD * NTOK * NTOK, lscale + d * NHEAD,
        ybuf, shift);
    gemm_bf16_kernel<0><<<dim3(C_DIM / 128, MB), 256, 0, stream>>>(
        ybuf, wproj + (size_t)d * C_DIM * C_DIM, proj_b + d * C_DIM, tbuf,
        M_TOK, C_DIM, C_DIM);
    add_ln_kernel<<<M_TOK, 256, 0, stream>>>(xbuf, tbuf, norm1_g + d * C_DIM,
                                             norm1_b + d * C_DIM);
    gemm_bf16_kernel<1><<<dim3((4 * C_DIM) / 128, MB), 256, 0, stream>>>(
        xbuf, wfc1 + (size_t)d * 4 * C_DIM * C_DIM, fc1_b + d * 4 * C_DIM,
        hbuf, M_TOK, 4 * C_DIM, C_DIM);
    gemm_bf16_kernel<0><<<dim3(C_DIM / 128, MB), 256, 0, stream>>>(
        hbuf, wfc2 + (size_t)d * C_DIM * 4 * C_DIM, fc2_b + d * C_DIM, tbuf,
        M_TOK, C_DIM, 4 * C_DIM);
    add_ln_kernel<<<M_TOK, 256, 0, stream>>>(xbuf, tbuf, norm2_g + d * C_DIM,
                                             norm2_b + d * C_DIM);
  }

  fuse_pack_kernel<<<(M_FUSE * 3 * C_DIM + 255) / 256, 256, 0, stream>>>(xbuf,
                                                                         ybuf);
  gemm_bf16_kernel<0><<<dim3(C_DIM / 128, (M_FUSE + 127) / 128), 256, 0,
                        stream>>>(ybuf, wfus, fus_b, hbuf, M_FUSE, C_DIM,
                                  3 * C_DIM);
  ln_out_kernel<<<M_FUSE, 256, 0, stream>>>(hbuf, fn_g, fn_b, (float*)d_out);
}
